// GCNLayer_48189533061406
// MI455X (gfx1250) — matrix operand order repaired and checked by reading, not compiled, not run on hardware
//
#include <hip/hip_runtime.h>

// Problem dimensions (fixed by the reference).
#define NNODES 30000
#define NEDGES 480000
#define CIN    1024
#define COUT   512
#define MTILES (NNODES / 16)        // 1875 (exact)
#define KSTEPS (CIN / 32)           // 32  (exact)

typedef __bf16 bf16_t;
typedef bf16_t v16bf __attribute__((ext_vector_type(16)));
typedef bf16_t v8bf  __attribute__((ext_vector_type(8)));
typedef float  v8f   __attribute__((ext_vector_type(8)));

#if __has_builtin(__builtin_amdgcn_tensor_load_to_lds) && \
    __has_builtin(__builtin_amdgcn_s_wait_tensorcnt)
#define HAVE_TDM 1
typedef unsigned int v4u __attribute__((ext_vector_type(4)));
typedef int          v8i __attribute__((ext_vector_type(8)));
typedef int          v4i __attribute__((ext_vector_type(4)));
#endif

// f32 -> bf16 RNE bits (used only in the one-shot conversion pre-pass).
__device__ __forceinline__ unsigned short bfbits(float f) {
    unsigned u = __builtin_bit_cast(unsigned, f);
    u += 0x7FFFu + ((u >> 16) & 1u);
    return (unsigned short)(u >> 16);
}

// ---------------------------------------------------------------------------
// Pre-pass: convert an f32 array to bf16 (float4 in, 4xbf16 = 8B out).
// Pure bandwidth; runs once over x (123MB) and W (2MB).
// ---------------------------------------------------------------------------
__global__ __launch_bounds__(256)
void f32_to_bf16_kernel(const float* __restrict__ src,
                        unsigned short* __restrict__ dst, int n4)
{
    int i = blockIdx.x * blockDim.x + threadIdx.x;
    if (i >= n4) return;
    float4 f = ((const float4*)src)[i];
    ushort4 o;
    o.x = bfbits(f.x); o.y = bfbits(f.y); o.z = bfbits(f.z); o.w = bfbits(f.w);
    ((ushort4*)dst)[i] = o;
}

#ifdef HAVE_TDM
// Issue a TDM load of a 64-row x 32-col bf16 W tile (row stride CIN elements)
// into LDS. D# layout per CDNA5 ISA (CDNA5 ISA):
//   g0: [1:0]=count=1, [63:32]=lds_addr, [120:64]=global_addr, [127:126]=type=2
//   g1: data_size=1(2B), tensor_dim0=CIN, tensor_dim1=COUT,
//       tile_dim0=32, tile_dim1=64, tensor_dim0_stride=CIN
// This toolchain exposes the 6-arg builtin:
//   (uint32x4 g0, int32x8 g1, int32x4, int32x4, int32x8, i32 cpol)
__device__ __forceinline__ void tdm_load_w_tile(const bf16_t* gsrc,
                                                unsigned ldsByteAddr)
{
    unsigned long long ga = (unsigned long long)(size_t)gsrc;
    v4u g0;
    g0[0] = 1u;                                         // count=1 (valid)
    g0[1] = ldsByteAddr;                                // lds_addr
    g0[2] = (unsigned)(ga & 0xFFFFFFFFu);               // global_addr[31:0]
    g0[3] = (unsigned)((ga >> 32) & 0x01FFFFFFu)        // global_addr[56:32]
          | (2u << 30);                                 // type=2 ("image")
    v8i g1;
    g1[0] = 0x10000;              // workgroup_mask=0, data_size=1 (2 bytes)
    g1[1] = (int)((unsigned)CIN << 16);    // tensor_dim0[15:0]=1024
    g1[2] = (int)((unsigned)COUT << 16);   // tensor_dim1[15:0]=512
    g1[3] = (int)(32u << 16);              // tile_dim0=32
    g1[4] = 64;                            // tile_dim1=64, tile_dim2=0 (2D)
    g1[5] = CIN;                           // tensor_dim0_stride=1024
    g1[6] = 0;
    g1[7] = 0;
    v4i z4 = {0, 0, 0, 0};                 // 2D tensor: groups 2/3 unused
    v8i z8 = {0, 0, 0, 0, 0, 0, 0, 0};     // unused trailing group
    __builtin_amdgcn_tensor_load_to_lds(g0, g1, z4, z4, z8, 0);
}
#endif

// ---------------------------------------------------------------------------
// GEMM: h[30000][512] = xbf[30000][1024] * Wbf[512][1024]^T (bf16 WMMA, f32 acc)
//
// Block = 128 threads = 4 waves sharing one 64-col N group. Per 32-wide K step
// the 64x32 bf16 W tile lands in LDS -- via the Tensor Data Mover (double-
// buffered, wave 0 issues, s_wait_tensorcnt + workgroup barrier) when the TDM
// builtin exists, else via cooperative b128 copies. Each wave then runs 4
// v_wmma_f32_16x16x32_bf16 against its global-loaded A fragment.
// ---------------------------------------------------------------------------
__global__ __launch_bounds__(128)
void gcn_gemm_bf16(const bf16_t* __restrict__ xbf,
                   const bf16_t* __restrict__ Wbf,
                   float* __restrict__ h)
{
    __shared__ alignas(64) bf16_t tileW[2][64 * 32];    // 2 x 4KB double buffer

    const int lane    = threadIdx.x & 31;
    const int wave    = threadIdx.x >> 5;
    const int halfSel = lane >> 4;        // 0: lanes 0-15, 1: lanes 16-31
    const int lrow    = lane & 15;

    int mTile = blockIdx.x * 4 + wave;
    if (mTile >= MTILES) mTile = MTILES - 1;  // duplicate last tile: keeps EXEC
                                              // all-ones + barriers uniform
    const int nBase = blockIdx.y * 64;

    const bf16_t* xrow  = xbf + (size_t)(mTile * 16 + lrow) * CIN;
    const bf16_t* wbase = Wbf + (size_t)nBase * CIN;    // row nBase, k=0

    v8f acc[4] = {};

#ifdef HAVE_TDM
    const unsigned lds0 = (unsigned)(size_t)(void*)&tileW[0][0]; // LDS byte addr
    const unsigned lds1 = (unsigned)(size_t)(void*)&tileW[1][0];
    // Prologue: DMA tile 0, wait, publish.
    if (wave == 0) {
        tdm_load_w_tile(wbase, lds0);
        __builtin_amdgcn_s_wait_tensorcnt(0);
    }
    __syncthreads();
#else
    // Fallback staging map: thread -> (row 0..63, 16-elem k segment)
    const int     trow = threadIdx.x >> 1;          // 0..63
    const int     tk   = (threadIdx.x & 1) * 16;    // 0 or 16
    const bf16_t* wsrc = Wbf + (size_t)(nBase + trow) * CIN + tk;
    bf16_t*       wst  = &tileW[0][trow * 32 + tk];
#endif

    for (int ks = 0; ks < KSTEPS; ++ks) {
        const int k   = ks * 32;
        const int cur = ks & 1;

#ifdef HAVE_TDM
        // Overlap: DMA next W tile into the other buffer while computing.
        if (wave == 0 && (ks + 1) < KSTEPS) {
            tdm_load_w_tile(wbase + (k + 32), (cur ? lds0 : lds1));
        }
#else
        *(v8bf*)wst       = *(const v8bf*)(wsrc + k);
        *(v8bf*)(wst + 8) = *(const v8bf*)(wsrc + k + 8);
        __syncthreads();
#endif

        // A fragment (contiguous k order, same as B): two contiguous 16B bf16 runs.
        //   lanes 0-15:  K = k+{0..15}
        //   lanes 16-31: K = k+{16..31}
        const int k0  = k + halfSel * 16;
        v8bf alo = *(const v8bf*)(xrow + k0);
        v8bf ahi = *(const v8bf*)(xrow + k0 + 8);
        v16bf afrag = __builtin_shufflevector(
            alo, ahi, 0, 1, 2, 3, 4, 5, 6, 7, 8, 9, 10, 11, 12, 13, 14, 15);

        // B fragments from LDS (contiguous k order as in A: lane = col n, lanes 0-15 hold
        // K=k..k+15, lanes 16-31 hold K=k+16..k+31): one 32B LDS read each.
#ifdef HAVE_TDM
        const bf16_t* wtile = &tileW[cur][0];
#else
        const bf16_t* wtile = &tileW[0][0];
#endif
#pragma unroll
        for (int t = 0; t < 4; ++t) {
            const v16bf bfrag =
                *(const v16bf*)&wtile[(t * 16 + lrow) * 32 + halfSel * 16];
            acc[t] = __builtin_amdgcn_wmma_f32_16x16x32_bf16(
                false, afrag, false, bfrag, (short)0, acc[t], false, false);
        }

#ifdef HAVE_TDM
        if (wave == 0) __builtin_amdgcn_s_wait_tensorcnt(0);
#endif
        __syncthreads();   // next-tile data published / tileW reusable
    }

    // Store D tiles: VGPR r -> row r (lanes 0-15) / r+8 (lanes 16-31), col lane&15.
    float* hp = h + (size_t)(mTile * 16 + halfSel * 8) * COUT + nBase + lrow;
#pragma unroll
    for (int t = 0; t < 4; ++t) {
#pragma unroll
        for (int r = 0; r < 8; ++r) {
            hp[(size_t)r * COUT + t * 16] = acc[t][r];
        }
    }
}

// ---------------------------------------------------------------------------
// SpMM scatter: out[row] += val * h[col]. One block per edge; 256 threads x
// float2 cover 512 channels. h and out both fit in the 192MB L2, so the f32
// atomics stream at L2 bandwidth.
// ---------------------------------------------------------------------------
__global__ __launch_bounds__(256)
void gcn_spmm_scatter(const long long* __restrict__ ei,   // [2][NEDGES] int64
                      const float* __restrict__ vals,
                      const float* __restrict__ h,
                      float* __restrict__ out)
{
    const int   e = blockIdx.x;
    const int   r = (int)ei[e];            // row = edge_index[0][e]
    const int   c = (int)ei[NEDGES + e];   // col = edge_index[1][e]
    const float v = vals[e];

    const int    t = threadIdx.x;          // 0..255 -> 2 channels each
    const float2 m = *(const float2*)(h + (size_t)c * COUT + t * 2);
    float*      op = out + (size_t)r * COUT + t * 2;
    unsafeAtomicAdd(op,     v * m.x);      // global_atomic_add_f32
    unsafeAtomicAdd(op + 1, v * m.y);
}

// ---------------------------------------------------------------------------
extern "C" void kernel_launch(void* const* d_in, const int* in_sizes, int n_in,
                              void* d_out, int out_size, void* d_ws, size_t ws_size,
                              hipStream_t stream)
{
    (void)in_sizes; (void)n_in; (void)ws_size;

    const float*     x  = (const float*)d_in[0];       // [30000][1024]
    const long long* ei = (const long long*)d_in[1];   // [2][480000] int64
    const float*     av = (const float*)d_in[2];       // [480000]
    const float*     W  = (const float*)d_in[3];       // [512][1024]
    float* out = (float*)d_out;                        // [30000][512]

    // Workspace layout (needs ~124MB):
    //   [0,                 61.44MB)  h   : f32 [30000][512]
    //   [61.44MB,          122.88MB)  xbf : bf16 [30000][1024]
    //   [122.88MB,         123.93MB)  Wbf : bf16 [512][1024]
    const size_t HB = (size_t)NNODES * COUT * sizeof(float);
    const size_t XB = (size_t)NNODES * CIN * sizeof(unsigned short);
    float*  h   = (float*)d_ws;
    bf16_t* xbf = (bf16_t*)((char*)d_ws + HB);
    bf16_t* wbf = (bf16_t*)((char*)d_ws + HB + XB);

    // out is poisoned by the harness; segment-sum needs zeros.
    (void)hipMemsetAsync(d_out, 0, (size_t)out_size * sizeof(float), stream);

    // One-shot f32 -> bf16 pre-pass (bandwidth-bound, halves GEMM read bytes).
    const int xn4 = NNODES * CIN / 4;      // 7,680,000
    const int wn4 = COUT * CIN / 4;        //   131,072
    f32_to_bf16_kernel<<<(xn4 + 255) / 256, 256, 0, stream>>>(
        x, (unsigned short*)xbf, xn4);
    f32_to_bf16_kernel<<<(wn4 + 255) / 256, 256, 0, stream>>>(
        W, (unsigned short*)wbf, wn4);

    dim3 gemmGrid((MTILES + 3) / 4, COUT / 64, 1);     // 469 x 8
    gcn_gemm_bf16<<<gemmGrid, 128, 0, stream>>>(xbf, wbf, h);

    gcn_spmm_scatter<<<NEDGES, 256, 0, stream>>>(ei, av, h, out);
}
